// MultiHeadAttention_12472585028194
// MI455X (gfx1250) — compile-verified, matrix operand order checked by reading, not run on hardware
//
#include <hip/hip_runtime.h>

// ---------------------------------------------------------------------------
// MHA forward for gfx1250 (CDNA5): bf16 WMMA GEMMs, fused attention.
// B=4, S=2048, D=1024, H=16, DEPTH=64
// d_out = [ out (B*S*D fp32) | attn (B*H*S*S fp32) ]
// d_ws  = q_bf16 | k_bf16 | vT_bf16 | ctx_bf16  (4 x 16.78 MB)
//
// Pipeline:
//   proj x3 (WMMA GEMM, fp32->bf16 LDS staging)
//   fused attention (scores WMMA -> online max -> exp/sum + ctx WMMA -> attn)
//   outproj (WMMA GEMM, async-to-LDS staging of bf16 A tiles)
// ---------------------------------------------------------------------------

#define BB 4
#define SS 2048
#define DD 1024
#define HH 16
#define DH 64

typedef __bf16 v16bf __attribute__((ext_vector_type(16)));
typedef __bf16 bf16x8 __attribute__((ext_vector_type(8)));
typedef float  v8f   __attribute__((ext_vector_type(8)));

__device__ __forceinline__ v8f wmma_bf16(v16bf a, v16bf b, v8f c) {
  return __builtin_amdgcn_wmma_f32_16x16x32_bf16(
      /*neg_a=*/false, a, /*neg_b=*/false, b,
      /*c_mod=*/(short)0, c, /*reuse_a=*/false, /*reuse_b=*/false);
}

__device__ __forceinline__ v16bf make_frag(bf16x8 lo, bf16x8 hi) {
  v16bf f;
#pragma unroll
  for (int i = 0; i < 8; ++i) { f[i] = lo[i]; f[i + 8] = hi[i]; }
  return f;
}

// load a 16x32 K-contiguous fragment: elements 0..7 = K kb..kb+7,
// elements 8..15 = K kb+16..kb+23 (kb = lane<16 ? 0 : 8), row = lane&15.
__device__ __forceinline__ v16bf load_frag(const __bf16* rowp, int kb) {
  return make_frag(*(const bf16x8*)(rowp + kb),
                   *(const bf16x8*)(rowp + kb + 16));
}

// ---------------------------------------------------------------------------
// Kernel 1: projection GEMM.  C[8192,1024] = X[8192,1024] @ W[1024,1024] + bias
// vmode==0: bf16 out[b][h][s][d] (q, k);  vmode==1: bf16 out[b][h][d][s] (v).
// block 256 = 8 waves, block tile 128(M) x 64(N), K-step 32.
// ---------------------------------------------------------------------------
__global__ void __launch_bounds__(256)
mha_proj_kernel(const float* __restrict__ X, const float* __restrict__ W,
                const float* __restrict__ bias, __bf16* __restrict__ out,
                int vmode) {
  __shared__ alignas(16) __bf16 As[128][32];
  __shared__ alignas(16) __bf16 Bs[64][32];

  const int m0 = blockIdx.x * 128;
  const int n0 = blockIdx.y * 64;
  const int tid = threadIdx.x;
  const int wave = tid >> 5;
  const int lane = tid & 31;
  const int lr = lane & 15;
  const int kb = (lane < 16) ? 0 : 8;

  v8f acc[4] = {};

  for (int k0 = 0; k0 < DD; k0 += 32) {
    // ---- batch all global loads first (overlap latencies) ----
    float4 fa[4];
#pragma unroll
    for (int i = 0; i < 4; ++i) {
      int linear = tid + i * 256;            // 1024 float4 chunks of A
      int row = linear >> 3;
      int c4 = (linear & 7) << 2;
      fa[i] = *(const float4*)(X + (size_t)(m0 + row) * DD + k0 + c4);
    }
    float4 fb[2];
#pragma unroll
    for (int i = 0; i < 2; ++i) {
      int linear = tid + i * 256;            // 512 float4 chunks of W
      int kr = linear >> 4;
      int c4 = (linear & 15) << 2;
      fb[i] = *(const float4*)(W + (size_t)(k0 + kr) * DD + n0 + c4);
    }
    if (k0 + 32 < DD)
      __builtin_prefetch(X + (size_t)(m0 + (tid >> 3)) * DD + k0 + 32, 0, 1);

    // ---- convert + store to LDS ----
#pragma unroll
    for (int i = 0; i < 4; ++i) {
      int linear = tid + i * 256;
      int row = linear >> 3;
      int c4 = (linear & 7) << 2;
      As[row][c4 + 0] = (__bf16)fa[i].x; As[row][c4 + 1] = (__bf16)fa[i].y;
      As[row][c4 + 2] = (__bf16)fa[i].z; As[row][c4 + 3] = (__bf16)fa[i].w;
    }
#pragma unroll
    for (int i = 0; i < 2; ++i) {
      int linear = tid + i * 256;
      int kr = linear >> 4;
      int c4 = (linear & 15) << 2;
      Bs[c4 + 0][kr] = (__bf16)fb[i].x; Bs[c4 + 1][kr] = (__bf16)fb[i].y;
      Bs[c4 + 2][kr] = (__bf16)fb[i].z; Bs[c4 + 3][kr] = (__bf16)fb[i].w;
    }
    __syncthreads();

    // ---- fragments: hoist all loads, then back-to-back WMMAs ----
    v16bf a = load_frag(&As[wave * 16 + lr][0], kb);
    v16bf bfr[4];
#pragma unroll
    for (int nt = 0; nt < 4; ++nt) bfr[nt] = load_frag(&Bs[nt * 16 + lr][0], kb);
#pragma unroll
    for (int nt = 0; nt < 4; ++nt) acc[nt] = wmma_bf16(a, bfr[nt], acc[nt]);
    __syncthreads();
  }

#pragma unroll
  for (int nt = 0; nt < 4; ++nt) {
    int n = n0 + nt * 16 + lr;
    int h = n >> 6;
    int d = n & 63;
    float bv = bias[n];
#pragma unroll
    for (int r = 0; r < 8; ++r) {
      int m = m0 + wave * 16 + ((lane < 16) ? r : (r + 8));
      int bidx = m >> 11;
      int s = m & 2047;
      float val = acc[nt][r] + bv;
      size_t idx;
      if (vmode)
        idx = ((size_t)(bidx * HH + h) * DH + d) * SS + s;   // [b][h][d][s]
      else
        idx = ((size_t)(bidx * HH + h) * SS + s) * DH + d;   // [b][h][s][d]
      out[idx] = (__bf16)val;
    }
  }
}

// ---------------------------------------------------------------------------
// Kernel 2: fused attention.  Per (bh, 128-row block): each wave owns 16 rows.
//   P1: scores via WMMA, per-lane running max -> shfl-xor reduce.
//   P2: recompute scores, e=exp(s-m), row-sum, LDS round-trip (C-layout ->
//       A-fragment), ctx WMMA accumulate with unnormalized e.
//   P3: recompute scores, write attn = exp(s-m)/l (fp32, 1 GB write-only).
//   Finally write ctx = O/l as bf16 [b][s][h*64+d].
// q fragments live in registers for all phases; k/v stay L2-resident.
// ---------------------------------------------------------------------------
__global__ void __launch_bounds__(256)
mha_attn_fused_kernel(const __bf16* __restrict__ q, const __bf16* __restrict__ k,
                      const __bf16* __restrict__ vT, float* __restrict__ attn,
                      __bf16* __restrict__ ctx) {
  __shared__ alignas(16) __bf16 Ew[8][16][32];   // per-wave e-tile (8 KB)

  const int tid = threadIdx.x;
  const int wave = tid >> 5;
  const int lane = tid & 31;
  const int lr = lane & 15;
  const int kb = (lane < 16) ? 0 : 8;
  const int bh = blockIdx.z;
  const int bidx = bh >> 4;
  const int h = bh & 15;
  const int m0 = blockIdx.x * 128 + wave * 16;
  const float scale = 0.125f;  // 1/sqrt(64)

  const __bf16* qrow = q + ((size_t)bh * SS + (m0 + lr)) * DH;
  v16bf a0 = load_frag(qrow, kb);        // K = 0..31
  v16bf a1 = load_frag(qrow + 32, kb);   // K = 32..63
  const __bf16* kbase = k + (size_t)bh * SS * DH;
  const __bf16* vbase = vT + (size_t)bh * DH * SS;

  // ---------------- P1: row max ----------------
  float rm[8];
#pragma unroll
  for (int r = 0; r < 8; ++r) rm[r] = -INFINITY;
  for (int n0 = 0; n0 < SS; n0 += 16) {
    const __bf16* krow = kbase + (size_t)(n0 + lr) * DH;
    v16bf b0 = load_frag(krow, kb);
    v16bf b1 = load_frag(krow + 32, kb);
    v8f acc = {};
    acc = wmma_bf16(a0, b0, acc);
    acc = wmma_bf16(a1, b1, acc);
#pragma unroll
    for (int r = 0; r < 8; ++r) rm[r] = fmaxf(rm[r], acc[r] * scale);
  }
  // reduce across the 16 lanes of each half (rows are disjoint per half)
#pragma unroll
  for (int mask = 1; mask < 16; mask <<= 1)
#pragma unroll
    for (int r = 0; r < 8; ++r) rm[r] = fmaxf(rm[r], __shfl_xor(rm[r], mask, 32));

  // ---------------- P2: row sum + ctx accumulate ----------------
  float rl[8] = {};
  v8f O[4] = {};
  for (int n0 = 0; n0 < SS; n0 += 32) {
#pragma unroll
    for (int t = 0; t < 2; ++t) {
      const __bf16* krow = kbase + (size_t)(n0 + t * 16 + lr) * DH;
      v16bf b0 = load_frag(krow, kb);
      v16bf b1 = load_frag(krow + 32, kb);
      v8f acc = {};
      acc = wmma_bf16(a0, b0, acc);
      acc = wmma_bf16(a1, b1, acc);
#pragma unroll
      for (int r = 0; r < 8; ++r) {
        float e = __expf(acc[r] * scale - rm[r]);
        rl[r] += e;
        int row = (lane < 16) ? r : (r + 8);
        Ew[wave][row][t * 16 + lr] = (__bf16)e;   // C-layout -> LDS
      }
    }
    // same-wave LDS ops are in-order: safe to read the tile back as A-frag
    v16bf ea = load_frag(&Ew[wave][lr][0], kb);
    v16bf bv[4];
#pragma unroll
    for (int dt = 0; dt < 4; ++dt)
      bv[dt] = load_frag(vbase + (size_t)(dt * 16 + lr) * SS + n0, kb);
#pragma unroll
    for (int dt = 0; dt < 4; ++dt) O[dt] = wmma_bf16(ea, bv[dt], O[dt]);
  }
#pragma unroll
  for (int mask = 1; mask < 16; mask <<= 1)
#pragma unroll
    for (int r = 0; r < 8; ++r) rl[r] += __shfl_xor(rl[r], mask, 32);
  float invl[8];
#pragma unroll
  for (int r = 0; r < 8; ++r) invl[r] = 1.f / rl[r];

  // ---------------- P3: write normalized attn ----------------
  float* abase = attn + (size_t)bh * SS * SS;
  for (int n0 = 0; n0 < SS; n0 += 16) {
    const __bf16* krow = kbase + (size_t)(n0 + lr) * DH;
    v16bf b0 = load_frag(krow, kb);
    v16bf b1 = load_frag(krow + 32, kb);
    v8f acc = {};
    acc = wmma_bf16(a0, b0, acc);
    acc = wmma_bf16(a1, b1, acc);
    int n = n0 + lr;
#pragma unroll
    for (int r = 0; r < 8; ++r) {
      int m = m0 + ((lane < 16) ? r : (r + 8));
      abase[(size_t)m * SS + n] = __expf(acc[r] * scale - rm[r]) * invl[r];
    }
  }

  // ---------------- write ctx (normalize O) ----------------
#pragma unroll
  for (int dt = 0; dt < 4; ++dt) {
    int d = dt * 16 + lr;
#pragma unroll
    for (int r = 0; r < 8; ++r) {
      int s = m0 + ((lane < 16) ? r : (r + 8));
      ctx[((size_t)bidx * SS + s) * DD + h * DH + d] = (__bf16)(O[dt][r] * invl[r]);
    }
  }
}

// ---------------------------------------------------------------------------
// Kernel 3: out = ctx(bf16)[8192,1024] @ WO[1024,1024] + bias -> fp32.
// A tile is a pure bf16 copy -> use CDNA5 async global->LDS DMA (ASYNCcnt).
// ---------------------------------------------------------------------------
__global__ void __launch_bounds__(256)
mha_outproj_kernel(const __bf16* __restrict__ A, const float* __restrict__ W,
                   const float* __restrict__ bias, float* __restrict__ out) {
  __shared__ alignas(16) __bf16 As[128][32];
  __shared__ alignas(16) __bf16 Bs[64][32];

  const int m0 = blockIdx.x * 128;
  const int n0 = blockIdx.y * 64;
  const int tid = threadIdx.x;
  const int wave = tid >> 5;
  const int lane = tid & 31;
  const int lr = lane & 15;
  const int kb = (lane < 16) ? 0 : 8;

  v8f acc[4] = {};

  for (int k0 = 0; k0 < DD; k0 += 32) {
    // ---- A tile: async copy global->LDS, 512 x 16B chunks ----
#pragma unroll
    for (int i = 0; i < 2; ++i) {
      int linear = tid + i * 256;
      int row = linear >> 2;
      int c8 = (linear & 3) << 3;
      unsigned lds_off =
          (unsigned)(uintptr_t)&As[row][c8];
      unsigned long long gaddr =
          (unsigned long long)(uintptr_t)(A + (size_t)(m0 + row) * DD + k0 + c8);
      asm volatile("global_load_async_to_lds_b128 %0, %1, off"
                   :: "v"(lds_off), "v"(gaddr)
                   : "memory");
    }
    // ---- B tile: fp32 -> bf16, transposed to Bs[n][k] ----
    float4 fb[2];
#pragma unroll
    for (int i = 0; i < 2; ++i) {
      int linear = tid + i * 256;
      int kr = linear >> 4;
      int c4 = (linear & 15) << 2;
      fb[i] = *(const float4*)(W + (size_t)(k0 + kr) * DD + n0 + c4);
    }
#pragma unroll
    for (int i = 0; i < 2; ++i) {
      int linear = tid + i * 256;
      int kr = linear >> 4;
      int c4 = (linear & 15) << 2;
      Bs[c4 + 0][kr] = (__bf16)fb[i].x; Bs[c4 + 1][kr] = (__bf16)fb[i].y;
      Bs[c4 + 2][kr] = (__bf16)fb[i].z; Bs[c4 + 3][kr] = (__bf16)fb[i].w;
    }
    asm volatile("s_wait_asynccnt 0" ::: "memory");
    __syncthreads();

    v16bf a = load_frag(&As[wave * 16 + lr][0], kb);
    v16bf bfr[4];
#pragma unroll
    for (int nt = 0; nt < 4; ++nt) bfr[nt] = load_frag(&Bs[nt * 16 + lr][0], kb);
#pragma unroll
    for (int nt = 0; nt < 4; ++nt) acc[nt] = wmma_bf16(a, bfr[nt], acc[nt]);
    __syncthreads();
  }

#pragma unroll
  for (int nt = 0; nt < 4; ++nt) {
    int n = n0 + nt * 16 + lr;
    float bv = bias[n];
#pragma unroll
    for (int r = 0; r < 8; ++r) {
      int m = m0 + wave * 16 + ((lane < 16) ? r : (r + 8));
      out[(size_t)m * DD + n] = acc[nt][r] + bv;
    }
  }
}

// ---------------------------------------------------------------------------
extern "C" void kernel_launch(void* const* d_in, const int* in_sizes, int n_in,
                              void* d_out, int out_size, void* d_ws,
                              size_t ws_size, hipStream_t stream) {
  const float* Q   = (const float*)d_in[0];
  const float* K   = (const float*)d_in[1];
  const float* V   = (const float*)d_in[2];
  const float* WQw = (const float*)d_in[3];
  const float* WQb = (const float*)d_in[4];
  const float* WKw = (const float*)d_in[5];
  const float* WKb = (const float*)d_in[6];
  const float* WVw = (const float*)d_in[7];
  const float* WVb = (const float*)d_in[8];
  const float* WOw = (const float*)d_in[9];
  const float* WOb = (const float*)d_in[10];

  const size_t nqkv = (size_t)BB * HH * SS * DH;  // 8,388,608 elems
  __bf16* q_bf   = (__bf16*)d_ws;
  __bf16* k_bf   = q_bf + nqkv;
  __bf16* vT_bf  = k_bf + nqkv;
  __bf16* ctx_bf = vT_bf + nqkv;                  // [B][S][D]

  float* out  = (float*)d_out;
  float* attn = out + (size_t)BB * SS * DD;       // [B*H][S][S]

  dim3 blk(256);
  dim3 gproj(8192 / 128, 1024 / 64);              // (64, 16)

  mha_proj_kernel<<<gproj, blk, 0, stream>>>(Q, WQw, WQb, q_bf, 0);
  mha_proj_kernel<<<gproj, blk, 0, stream>>>(K, WKw, WKb, k_bf, 0);
  mha_proj_kernel<<<gproj, blk, 0, stream>>>(V, WVw, WVb, vT_bf, 1);

  mha_attn_fused_kernel<<<dim3(SS / 128, 1, BB * HH), blk, 0, stream>>>(
      q_bf, k_bf, vT_bf, attn, ctx_bf);

  mha_outproj_kernel<<<gproj, blk, 0, stream>>>(ctx_bf, WOw, WOb, out);
}
